// ConvCapsuleLayer_90735479095913
// MI455X (gfx1250) — matrix operand order repaired and checked by reading, not compiled, not run on hardware
//
#include <hip/hip_runtime.h>
#include <math.h>

// CDNA5 WMMA vector types (wave32)
typedef __attribute__((ext_vector_type(16))) __bf16 v16bf;
typedef __attribute__((ext_vector_type(8)))  float  v8f;

#define NP_   8
#define LP_   16
#define NC_   4
#define B_    4
#define HH    256
#define WW    256
#define KTOT  400      // 16 in-ch * 5 * 5
#define KPAD  416      // 13 chunks of 32
#define NKC   13
#define TILE  16       // pixels per block (one row segment)
#define KSTR  432      // bl row stride in bf16 (864 B: 32B-aligned, 2-way banked)

// ---------------------------------------------------------------------------
// Kernel 0: pack W into bf16 A-fragments + build the K -> patch-offset table.
// A layout (16-bit A 16x32, contiguous K, same order as B): lanes 0-15 -> M rows, VGPR v half h:
//   K = 2v+h, lanes 16-31 add +16.
// ---------------------------------------------------------------------------
__global__ void pack_w_kernel(const float* __restrict__ W,
                              __bf16* __restrict__ wp,
                              int* __restrict__ tab) {
    int gid = blockIdx.x * blockDim.x + threadIdx.x;
    if (gid < NP_ * NKC * 32) {
        int mt   = gid / (NKC * 32);
        int kc   = (gid / 32) % NKC;
        int lane = gid & 31;
        int oc   = mt * 16 + (lane & 15);
        int khi  = (lane & 16) ? 16 : 0;
        __bf16* dst = wp + ((size_t)((mt * NKC + kc) * 32 + lane)) * 16;
#pragma unroll
        for (int i = 0; i < 16; ++i) {
            int v = i >> 1, hh = i & 1;
            int kk = i + khi;
            int K  = kc * 32 + kk;
            float val = (K < KTOT) ? W[oc * KTOT + K] : 0.0f;
            dst[i] = (__bf16)val;
        }
    } else if (gid < NP_ * NKC * 32 + KPAD) {
        int K = gid - NP_ * NKC * 32;
        int t = -1;
        if (K < KTOT) {
            int ic = K / 25;
            int r  = K % 25;
            int kh = r / 5;
            int kw = r - kh * 5;
            t = (ic * 5 + kh) * 20 + kw;   // offset into xs[ic][5][20]
        }
        tab[K] = t;
    }
}

__device__ __forceinline__ float red16(float v) {
#pragma unroll
    for (int o = 8; o > 0; o >>= 1) v += __shfl_xor(v, o, 16);
    return v;
}

// ---------------------------------------------------------------------------
// Kernel 1: fused implicit-GEMM conv (bf16 WMMA, f32 acc) + 3 routing iters.
// Block = 256 threads (8 waves) = one (b, h, 16-pixel) tile across all nc.
// Wave w owns M-tile w (== np w); keeps one f32 accumulator per nc.
// The nc loop is FULLY UNROLLED so acc[nc] is statically register-allocated
// (a rolled loop forced MOVREL-indexed VGPR copies around every WMMA).
// ---------------------------------------------------------------------------
__global__ __launch_bounds__(256) void caps_kernel(const float* __restrict__ x,
                                                   const __bf16* __restrict__ wp,
                                                   const int* __restrict__ tab,
                                                   float* __restrict__ out) {
    __shared__ float  xs[1792];          // patch: 16 ic x 5 rows x 20 cols (+pad)
    __shared__ __bf16 bl[TILE * KSTR];   // im2col B panel: 13.8 KB
    __shared__ float  vl[NC_ * NP_ * LP_ * TILE]; // votes: 32 KB
    __shared__ int    tabs[KPAD];

    const int tid = threadIdx.x;
    const int blk = blockIdx.x;
    const int w0  = (blk & 15) * TILE;
    const int h   = (blk >> 4) & 255;
    const int b   = blk >> 12;

    for (int i = tid; i < KPAD; i += 256) tabs[i] = tab[i];

    const int wid   = tid >> 5;          // wave id == mtile == np
    const int lane  = tid & 31;
    const int pix   = lane & 15;
    const int khalf = (lane & 16) ? 16 : 0;
    const __bf16* abase = wp + ((size_t)(wid * NKC * 32 + lane)) * 16;

    v8f acc[NC_] = {v8f{}, v8f{}, v8f{}, v8f{}};

#pragma unroll
    for (int nc = 0; nc < NC_; ++nc) {
        __syncthreads();   // xs/bl reuse + (first iter) tabs visibility
        // ---- stage this nc's input patch (zero-padded borders)
#pragma unroll
        for (int t = 0; t < 7; ++t) {
            int idx = tid + t * 256;
            if (idx < 1600) {
                int ic  = idx / 100;
                int row = (idx / 20) % 5;
                int col = idx % 20;
                int gh = h - 2 + row;
                int gw = w0 - 2 + col;
                float v = 0.0f;
                if (gh >= 0 && gh < HH && gw >= 0 && gw < WW)
                    v = x[(size_t)((b * NC_ + nc) * 16 + ic) * (HH * WW) + gh * WW + gw];
                xs[idx] = v;
            }
        }
        __syncthreads();
        // ---- im2col into WMMA-B layout: bl[pix][K] (bf16), 6656 elems / 256 thr
#pragma unroll
        for (int t = 0; t < 26; ++t) {
            int idx = tid + t * 256;     // idx = K*16 + pix
            int K   = idx >> 4;
            int px  = idx & 15;
            int off = tabs[K];
            float v = (off >= 0) ? xs[off + px] : 0.0f;
            bl[px * KSTR + K] = (__bf16)v;
        }
        __syncthreads();
        // ---- GEMM: 13 chunks of K=32, one WMMA each
#pragma unroll
        for (int kc = 0; kc < NKC; ++kc) {
            v16bf bfrag = *(const v16bf*)&bl[pix * KSTR + kc * 32 + khalf];
            v16bf afrag = *(const v16bf*)(abase + (size_t)kc * 512);
            acc[nc] = __builtin_amdgcn_wmma_f32_16x16x32_bf16(
                false, afrag, false, bfrag, (short)0, acc[nc], false, false);
        }
    }

    // ---- scatter D tiles into votes LDS: lane<16: M=r, N=lane; lane>=16: M=r+8
#pragma unroll
    for (int nc = 0; nc < NC_; ++nc) {
#pragma unroll
        for (int r = 0; r < 8; ++r) {
            int lp = r + ((lane & 16) ? 8 : 0);
            vl[((nc * NP_ + wid) * LP_ + lp) * TILE + pix] = acc[nc][r];
        }
    }
    __syncthreads();

    // ---- dynamic routing, 3 iterations. Team = 16 lanes per pixel.
    const int p = tid >> 4;      // pixel 0..15
    const int l = tid & 15;      // lp index

    float vv[NC_][NP_];
#pragma unroll
    for (int c = 0; c < NC_; ++c)
#pragma unroll
        for (int n = 0; n < NP_; ++n)
            vv[c][n] = vl[((c * NP_ + n) * LP_ + l) * TILE + p];

    float sims[NC_][NP_] = {};
    float parent[NP_];

#pragma unroll
    for (int it = 0; it < 3; ++it) {
        float cc[NC_][NP_];
#pragma unroll
        for (int c = 0; c < NC_; ++c) {
            float mx = sims[c][0];
#pragma unroll
            for (int n = 1; n < NP_; ++n) mx = fmaxf(mx, sims[c][n]);
            float s = 0.0f;
#pragma unroll
            for (int n = 0; n < NP_; ++n) { float e = __expf(sims[c][n] - mx); cc[c][n] = e; s += e; }
            float inv = 1.0f / s;
#pragma unroll
            for (int n = 0; n < NP_; ++n) cc[c][n] *= inv;
        }
#pragma unroll
        for (int n = 0; n < NP_; ++n) {
            float pb = 0.0f;
#pragma unroll
            for (int c = 0; c < NC_; ++c) pb += cc[c][n] * vv[c][n];
            float n2 = red16(pb * pb);
            float factor = __fsqrt_rn(n2) / (1.0f + n2 + 1e-4f);
            parent[n] = factor * pb;
        }
#pragma unroll
        for (int c = 0; c < NC_; ++c)
#pragma unroll
            for (int n = 0; n < NP_; ++n)
                sims[c][n] += red16(vv[c][n] * parent[n]);
    }

    // ---- write parent: out[b][np][lp=l][h][w0+p]
#pragma unroll
    for (int n = 0; n < NP_; ++n)
        out[(size_t)((b * NP_ + n) * LP_ + l) * (HH * WW) + h * WW + (w0 + p)] = parent[n];
}

extern "C" void kernel_launch(void* const* d_in, const int* in_sizes, int n_in,
                              void* d_out, int out_size, void* d_ws, size_t ws_size,
                              hipStream_t stream) {
    const float* x = (const float*)d_in[0];   // (4,4,16,256,256) f32
    const float* W = (const float*)d_in[1];   // (128,16,5,5) f32
    __bf16* wpack  = (__bf16*)d_ws;           // 8*13*32*16 bf16 = 104 KB
    int*    tab    = (int*)(wpack + NP_ * NKC * 32 * 16); // 416 ints

    // 8*13*32 = 3328 pack threads + 416 table threads = 3744 -> 15 blocks
    pack_w_kernel<<<15, 256, 0, stream>>>(W, wpack, tab);

    // blocks: 4 (B) * 256 (H) * 16 (W/16) = 16384
    caps_kernel<<<B_ * HH * (WW / TILE), 256, 0, stream>>>(x, wpack, tab, (float*)d_out);
}
